// Attention_90752658965090
// MI455X (gfx1250) — matrix operand order repaired and checked by reading, not compiled, not run on hardware
//
#include <hip/hip_runtime.h>

// ---------------------------------------------------------------------------
// Types
// ---------------------------------------------------------------------------
typedef _Float16 v16h __attribute__((ext_vector_type(16)));
typedef _Float16 h8   __attribute__((ext_vector_type(8)));
typedef _Float16 h4   __attribute__((ext_vector_type(4)));
typedef float    v8f  __attribute__((ext_vector_type(8)));
typedef unsigned int u32x4 __attribute__((ext_vector_type(4)));
typedef int      i32x8 __attribute__((ext_vector_type(8)));
typedef int      i32x4 __attribute__((ext_vector_type(4)));

// Problem constants
#define BATCH 16
#define CDIM  512
#define NDIM  4096
#define GROUPS 8

// GEMM tiling
#define BM 128
#define BN 128
#define BK 32
#define ASTR 48            // LDS A row stride (halves): 32 + 16 pad, 96B rows
#define BSTR 144           // LDS B row stride (halves): 128 + 16 pad, 288B rows
#define A_TILE (BM * ASTR) // halves per A buffer
#define B_TILE (BK * BSTR) // halves per B buffer

enum { MODE_QKV = 0, MODE_SCORES = 1, MODE_AV = 2, MODE_PROJ = 3 };

// TDM pad controls: pad_enable(bit20) | pad_interval[24:22] | pad_amount[31:25]
// interval code c -> 2^(c+1) DWORDs; amount code a -> (a+1) DWORDs
#define PADCTL_A ((1u << 20) | (3u << 22) | (7u << 25))  // every 16 DW, +8 DW -> stride 48 halves
#define PADCTL_B ((1u << 20) | (5u << 22) | (7u << 25))  // every 64 DW, +8 DW -> stride 144 halves

#if defined(__has_builtin)
#if __has_builtin(__builtin_amdgcn_tensor_load_to_lds) && \
    __has_builtin(__builtin_amdgcn_s_wait_tensorcnt) && defined(__gfx1250__)
#define USE_TDM 1
#endif
#endif
#ifndef USE_TDM
#define USE_TDM 0
#endif

// ---------------------------------------------------------------------------
// TDM: DMA a 2-D f16 tile (tile1 rows x tile0 elems, row stride stride0 elems)
// from global into LDS at lds_byte (packed rows + pad per padctl).
// D# layout per CDNA5 ISA.
// ---------------------------------------------------------------------------
__device__ __forceinline__ void tdm_load_2d(const _Float16* gsrc, unsigned lds_byte,
                                            int tdim0, int tdim1, int tile0, int tile1,
                                            long long stride0, unsigned padctl) {
#if USE_TDM
  unsigned long long ga = (unsigned long long)gsrc;
  u32x4 g0;
  g0[0] = 1u;                                   // count=1, user descriptor
  g0[1] = lds_byte;                             // lds_addr
  g0[2] = (unsigned)ga;                         // global_addr[31:0]
  g0[3] = ((unsigned)(ga >> 32) & 0x01FFFFFFu) | (2u << 30);  // addr[56:32] | type=2
  i32x8 g1;
  g1[0] = (int)((1u << 16) | padctl);           // wg_mask=0, data_size=2B, pad ctl
  g1[1] = (int)(((unsigned)tdim0 & 0xFFFFu) << 16);                 // tensor_dim0 lo
  g1[2] = (int)((((unsigned)tdim0 >> 16) & 0xFFFFu) |
                (((unsigned)tdim1 & 0xFFFFu) << 16));               // dim0 hi | dim1 lo
  g1[3] = (int)((((unsigned)tdim1 >> 16) & 0xFFFFu) |
                (((unsigned)tile0) << 16));                         // dim1 hi | tile_dim0
  g1[4] = (int)((unsigned)tile1);                                   // tile_dim1 | tile_dim2=0
  g1[5] = (int)(unsigned)(stride0 & 0xFFFFFFFFll);                  // dim0_stride lo
  g1[6] = (int)(unsigned)((stride0 >> 32) & 0xFFFFll);              // dim0_stride hi | dim1_stride lo
  g1[7] = 0;
  i32x4 z4 = {};
  i32x8 z8 = {};
  __builtin_amdgcn_tensor_load_to_lds(g0, g1, z4, z4, z8, 0);
#else
  (void)gsrc; (void)lds_byte; (void)tdim0; (void)tdim1;
  (void)tile0; (void)tile1; (void)stride0; (void)padctl;
#endif
}

// ---------------------------------------------------------------------------
// GroupNorm stats: per-(batch,group) mean / rstd over 64*4096 contiguous floats
// ---------------------------------------------------------------------------
__global__ __launch_bounds__(256) void gn_stats(const float* __restrict__ x,
                                                float* __restrict__ stats) {
  const int bg = blockIdx.x;
  const float4* p = (const float4*)(x + (size_t)bg * (64 * 4096));
  float s = 0.0f, q = 0.0f;
  for (int i = threadIdx.x; i < 65536; i += 256) {
    float4 v = p[i];
    s += v.x + v.y + v.z + v.w;
    q += v.x * v.x + v.y * v.y + v.z * v.z + v.w * v.w;
  }
  __shared__ float rs[256], rq[256];
  rs[threadIdx.x] = s;
  rq[threadIdx.x] = q;
  __syncthreads();
  for (int o = 128; o > 0; o >>= 1) {
    if (threadIdx.x < o) {
      rs[threadIdx.x] += rs[threadIdx.x + o];
      rq[threadIdx.x] += rq[threadIdx.x + o];
    }
    __syncthreads();
  }
  if (threadIdx.x == 0) {
    const float inv = 1.0f / 262144.0f;
    float mean = rs[0] * inv;
    float var  = rq[0] * inv - mean * mean;
    stats[2 * bg]     = mean;
    stats[2 * bg + 1] = rsqrtf(var + 1e-5f);
  }
}

__global__ __launch_bounds__(256) void gn_apply(const float* __restrict__ x,
                                                const float* __restrict__ stats,
                                                const float* __restrict__ gw,
                                                const float* __restrict__ gb,
                                                _Float16* __restrict__ xn) {
  const size_t total4 = (size_t)BATCH * CDIM * NDIM / 4;
  for (size_t i = (size_t)blockIdx.x * blockDim.x + threadIdx.x; i < total4;
       i += (size_t)gridDim.x * blockDim.x) {
    const size_t e = i * 4;
    const int c  = (int)((e >> 12) & 511);
    const int b  = (int)(e >> 21);
    const int sg = b * GROUPS + (c >> 6);
    const float mean = stats[2 * sg];
    const float rstd = stats[2 * sg + 1];
    const float a = gw[c] * rstd;
    const float d = gb[c] - mean * a;
    float4 v = ((const float4*)x)[i];
    h4 o;
    o[0] = (_Float16)(v.x * a + d);
    o[1] = (_Float16)(v.y * a + d);
    o[2] = (_Float16)(v.z * a + d);
    o[3] = (_Float16)(v.w * a + d);
    ((h4*)xn)[i] = o;
  }
}

__global__ __launch_bounds__(256) void f32_to_f16(const float* __restrict__ a,
                                                  _Float16* __restrict__ b, int n) {
  for (int i = blockIdx.x * blockDim.x + threadIdx.x; i < n;
       i += gridDim.x * blockDim.x)
    b[i] = (_Float16)a[i];
}

// ---------------------------------------------------------------------------
// WMMA GEMM: C[z] (MxN) = A (MxK rm, ld=K) * B
//   NN modes : B is K x N row-major; LDS keeps it row-major (B frag: lane = N
//              col, 16 K halves read down rows -> no transpose).
//   SCORES   : B is N x K row-major; staged transposed into LDS K x N.
// 256 thr = 8 waves; 128x128 tile; wave = 64x32 = 4x2 WMMA accums.
// Double-buffered: TDM (waves 0/1) / VALU staging of tile t+1 overlaps WMMA of t.
// ---------------------------------------------------------------------------
template <int MODE>
__global__ __launch_bounds__(256) void gemm_wmma(
    const _Float16* __restrict__ A, const _Float16* __restrict__ B,
    float* __restrict__ Cf, _Float16* __restrict__ Ch,
    const float* __restrict__ bias, const float* __restrict__ resid,
    float scale, int M, int N, int K,
    long long sA, long long sB, long long sC) {
  __shared__ _Float16 As[2 * A_TILE];
  __shared__ _Float16 Bs[2 * B_TILE];

  const int tid  = threadIdx.x;
  const int lane = tid & 31;
  const int wid  = tid >> 5;
  const int wm   = wid >> 2;      // 0..1 -> 64-row slab
  const int wn   = wid & 3;       // 0..3 -> 32-col slab

  const int bn = blockIdx.x * BN;
  const int bm = blockIdx.y * BM;
  const int z  = blockIdx.z;

  A += (size_t)z * (size_t)sA;
  B += (size_t)z * (size_t)sB;

  const unsigned ldsA = (unsigned)(unsigned long long)&As[0];
  const unsigned ldsB = (unsigned)(unsigned long long)&Bs[0];

  // stage tile (kt) into buffer q
  auto stage = [&](int q, int kt) {
#if USE_TDM
    if (wid == 0) {
      tdm_load_2d(A + (size_t)bm * K + kt, ldsA + (unsigned)(q * A_TILE * 2),
                  K - kt, M - bm, BK, BM, (long long)K, PADCTL_A);
    } else if (MODE != MODE_SCORES && wid == 1) {
      tdm_load_2d(B + (size_t)kt * N + bn, ldsB + (unsigned)(q * B_TILE * 2),
                  N - bn, K - kt, BN, BK, (long long)N, PADCTL_B);
    }
#else
    {
      _Float16* Asq = &As[q * A_TILE];
#pragma unroll
      for (int it = 0; it < 2; ++it) {
        int hh = (tid + it * 256) * 8;
        int r  = hh / BK;
        int cc = hh % BK;
        *(h8*)(&Asq[r * ASTR + cc]) =
            *(const h8*)(A + (size_t)(bm + r) * K + kt + cc);
      }
      if constexpr (MODE != MODE_SCORES) {
        _Float16* Bsq = &Bs[q * B_TILE];
#pragma unroll
        for (int it = 0; it < 2; ++it) {
          int hh = (tid + it * 256) * 8;
          int r  = hh / BN;
          int cc = hh % BN;
          *(h8*)(&Bsq[r * BSTR + cc]) =
              *(const h8*)(B + (size_t)(kt + r) * N + bn + cc);
        }
      }
    }
#endif
    if constexpr (MODE == MODE_SCORES) {
      // k-matrix is (N rows x K) row-major; transpose into LDS Bs[k][n]
      _Float16* Bsq = &Bs[q * B_TILE];
#pragma unroll
      for (int it = 0; it < 2; ++it) {
        int hh = (tid + it * 256) * 8;
        int r  = hh / BK;               // n row 0..127
        int cc = hh % BK;               // k col, multiple of 8
        h8 v = *(const h8*)(B + (size_t)(bn + r) * K + kt + cc);
#pragma unroll
        for (int j = 0; j < 8; ++j)
          Bsq[(cc + j) * BSTR + r] = v[j];
      }
    }
  };

  v8f acc[4][2] = {};

  // prologue
  stage(0, 0);
#if USE_TDM
  if (wid < 2) __builtin_amdgcn_s_wait_tensorcnt(0);
#endif
  __syncthreads();

  int p = 0;
  for (int kt = 0; kt < K; kt += BK, p ^= 1) {
    const bool has_next = (kt + BK) < K;
    if (has_next) stage(p ^ 1, kt + BK);

    // ---- fragments (ISA 7.12.2 layouts) ----
    const _Float16* Asq = &As[p * A_TILE];
    const _Float16* Bsq = &Bs[p * B_TILE];
    const int kj = lane >> 4;           // K-chunk select
    const int am = lane & 15;           // A: M row within 16

    v16h af[4], bf[2];
#pragma unroll
    for (int i = 0; i < 4; ++i) {
      const int row = (wm * 64 + i * 16 + am) * ASTR;
      h8 lo = *(const h8*)(&Asq[row + 8 * kj]);        // K = 8kj .. +8
      h8 hi = *(const h8*)(&Asq[row + 16 + 8 * kj]);   // K = 16+8kj .. +8
      af[i] = __builtin_shufflevector(lo, hi, 0, 1, 2, 3, 4, 5, 6, 7,
                                      8, 9, 10, 11, 12, 13, 14, 15);
    }
#pragma unroll
    for (int j = 0; j < 2; ++j)   // B: lane = N col (lane&15), K = 8kj+s / 16+8kj+(s-8)
#pragma unroll
      for (int s = 0; s < 16; ++s)
        bf[j][s] = Bsq[(s < 8 ? 8 * kj + s : 16 + 8 * kj + (s - 8)) * BSTR + wn * 32 + j * 16 + am];

#pragma unroll
    for (int i = 0; i < 4; ++i)
#pragma unroll
      for (int j = 0; j < 2; ++j)
        acc[i][j] = __builtin_amdgcn_wmma_f32_16x16x32_f16(
            false, af[i], false, bf[j], (short)0, acc[i][j], false, false);

#if USE_TDM
    if (has_next && wid < 2) __builtin_amdgcn_s_wait_tensorcnt(0);
#endif
    __syncthreads();
  }

  // ---- epilogue: C/D m = r + (lane>=16 ? 8 : 0), n = lane & 15 ----
  const int mofs = (lane >= 16) ? 8 : 0;
  const int ncol = lane & 15;
#pragma unroll
  for (int i = 0; i < 4; ++i) {
#pragma unroll
    for (int j = 0; j < 2; ++j) {
#pragma unroll
      for (int r = 0; r < 8; ++r) {
        const int m = bm + wm * 64 + i * 16 + r + mofs;
        const int n = bn + wn * 32 + j * 16 + ncol;
        const float v = acc[i][j][r];
        const size_t cidx = (size_t)z * (size_t)sC + (size_t)m * N + n;
        if constexpr (MODE == MODE_QKV) {
          Ch[cidx] = (_Float16)(v + bias[m]);
        } else if constexpr (MODE == MODE_SCORES) {
          Cf[cidx] = v * scale;
        } else if constexpr (MODE == MODE_AV) {
          Ch[cidx] = (_Float16)v;
        } else {
          Cf[cidx] = v + bias[m] + resid[cidx];
        }
      }
    }
  }
}

// ---------------------------------------------------------------------------
// Row softmax over (b, 512, 512) fp32 scores -> f16 probs
// ---------------------------------------------------------------------------
__global__ __launch_bounds__(256) void softmax_rows(const float* __restrict__ s,
                                                    _Float16* __restrict__ p) {
  const int row = blockIdx.x;
  const int b   = blockIdx.y;
  const size_t base = ((size_t)b * CDIM + row) * CDIM;
  const int t = threadIdx.x;
  __shared__ float red[256];

  const float v0 = s[base + t];
  const float v1 = s[base + t + 256];
  red[t] = fmaxf(v0, v1);
  __syncthreads();
  for (int o = 128; o > 0; o >>= 1) {
    if (t < o) red[t] = fmaxf(red[t], red[t + o]);
    __syncthreads();
  }
  const float mx = red[0];
  __syncthreads();

  const float e0 = __expf(v0 - mx);
  const float e1 = __expf(v1 - mx);
  red[t] = e0 + e1;
  __syncthreads();
  for (int o = 128; o > 0; o >>= 1) {
    if (t < o) red[t] += red[t + o];
    __syncthreads();
  }
  const float inv = 1.0f / red[0];
  p[base + t]       = (_Float16)(e0 * inv);
  p[base + t + 256] = (_Float16)(e1 * inv);
}

// ---------------------------------------------------------------------------
// Launch
// ---------------------------------------------------------------------------
extern "C" void kernel_launch(void* const* d_in, const int* in_sizes, int n_in,
                              void* d_out, int out_size, void* d_ws, size_t ws_size,
                              hipStream_t stream) {
  (void)in_sizes; (void)n_in; (void)out_size; (void)ws_size;

  const float* x      = (const float*)d_in[0];
  const float* norm_w = (const float*)d_in[1];
  const float* norm_b = (const float*)d_in[2];
  const float* qkv_w  = (const float*)d_in[3];
  const float* qkv_b  = (const float*)d_in[4];
  const float* proj_w = (const float*)d_in[5];
  const float* proj_b = (const float*)d_in[6];
  float* out = (float*)d_out;

  const int B = BATCH, C = CDIM, N = NDIM;

  char* ws = (char*)d_ws;
  size_t off = 0;
  auto alloc = [&](size_t bytes) -> void* {
    off = (off + 255) & ~((size_t)255);
    void* p = ws + off;
    off += bytes;
    return p;
  };
  float*    stats  = (float*)alloc((size_t)B * GROUPS * 2 * sizeof(float));
  _Float16* wqkvh  = (_Float16*)alloc((size_t)3 * C * C * sizeof(_Float16));
  _Float16* wprojh = (_Float16*)alloc((size_t)C * C * sizeof(_Float16));
  _Float16* xn     = (_Float16*)alloc((size_t)B * C * N * sizeof(_Float16));
  _Float16* qkv    = (_Float16*)alloc((size_t)B * 3 * C * N * sizeof(_Float16));
  float*    scores = (float*)alloc((size_t)B * C * C * sizeof(float));
  _Float16* probs  = (_Float16*)alloc((size_t)B * C * C * sizeof(_Float16));
  _Float16* attout = (_Float16*)alloc((size_t)B * C * N * sizeof(_Float16));

  gn_stats<<<dim3(B * GROUPS), dim3(256), 0, stream>>>(x, stats);
  gn_apply<<<dim3(8192), dim3(256), 0, stream>>>(x, stats, norm_w, norm_b, xn);

  f32_to_f16<<<dim3(768), dim3(256), 0, stream>>>(qkv_w, wqkvh, 3 * C * C);
  f32_to_f16<<<dim3(256), dim3(256), 0, stream>>>(proj_w, wprojh, C * C);

  gemm_wmma<MODE_QKV><<<dim3(N / BN, (3 * C) / BM, B), 256, 0, stream>>>(
      wqkvh, xn, nullptr, qkv, qkv_b, nullptr, 1.0f,
      3 * C, N, C, 0LL, (long long)C * N, (long long)3 * C * N);

  const float scale = 0.044194173824159216f;  // 512^-0.5
  gemm_wmma<MODE_SCORES><<<dim3(C / BN, C / BM, B), 256, 0, stream>>>(
      qkv, qkv + (size_t)C * N, scores, nullptr, nullptr, nullptr, scale,
      C, C, N, (long long)3 * C * N, (long long)3 * C * N, (long long)C * C);

  softmax_rows<<<dim3(C, B), 256, 0, stream>>>(scores, probs);

  gemm_wmma<MODE_AV><<<dim3(N / BN, C / BM, B), 256, 0, stream>>>(
      probs, qkv + (size_t)2 * C * N, nullptr, attout, nullptr, nullptr, 1.0f,
      C, N, C, (long long)C * C, (long long)3 * C * N, (long long)C * N);

  gemm_wmma<MODE_PROJ><<<dim3(N / BN, C / BM, B), 256, 0, stream>>>(
      wprojh, attout, out, nullptr, proj_b, x, 1.0f,
      C, N, C, 0LL, (long long)C * N, (long long)C * N);
}
